// SSMCore_56367150793445
// MI455X (gfx1250) — matrix operand order repaired and checked by reading, not compiled, not run on hardware
//
#include <hip/hip_runtime.h>

typedef _Float16 v16h __attribute__((ext_vector_type(16)));
typedef _Float16 v8h  __attribute__((ext_vector_type(8)));
typedef float    v8f  __attribute__((ext_vector_type(8)));

#define D_MODEL 2048
#define N_STATE 256
#define RANK    64
#define LEVELS  8
#define TOKENS  16384   /* B*S = 4*4096 */
#define TM      64      /* tokens per block (4 row groups of 16) */

// ----------------------------------------------------------------------------
// Prep kernel 1: build the 256x256 Kronecker ("butterfly") matrix M.
// state' = state @ M,  M[r,c] = prod_lvl A[lvl][bit(r,lvl), bit(c,lvl)],
// bit(n,lvl) = (n >> (7-lvl)) & 1  (level 0 acts on the MSB of the index).
// ----------------------------------------------------------------------------
__global__ void k_butterfly_M(const float* __restrict__ A, float* __restrict__ M) {
  int idx = blockIdx.x * blockDim.x + threadIdx.x;
  if (idx >= N_STATE * N_STATE) return;
  int r = idx >> 8, c = idx & 255;
  float v = 1.0f;
#pragma unroll
  for (int lvl = 0; lvl < LEVELS; ++lvl) {
    int i = (r >> (7 - lvl)) & 1;
    int j = (c >> (7 - lvl)) & 1;
    v *= A[lvl * 4 + i * 2 + j];
  }
  M[idx] = v;
}

// Prep kernel 2: Tm[256x64] = M @ W_c_rank
__global__ void k_m_wcrank(const float* __restrict__ M, const float* __restrict__ Wcr,
                           float* __restrict__ Tm) {
  int idx = blockIdx.x * blockDim.x + threadIdx.x;
  if (idx >= N_STATE * RANK) return;
  int r = idx >> 6, c = idx & 63;
  float s = 0.0f;
  for (int k = 0; k < N_STATE; ++k) s = fmaf(M[r * 256 + k], Wcr[k * 64 + c], s);
  Tm[idx] = s;
}

// Prep kernel 3: Wmid[64x64] = W_b_state @ Tm
__global__ void k_wbstate_tm(const float* __restrict__ Wbs, const float* __restrict__ Tm,
                             float* __restrict__ Wmid) {
  int idx = blockIdx.x * blockDim.x + threadIdx.x;
  if (idx >= RANK * RANK) return;
  int r = idx >> 6, c = idx & 63;
  float s = 0.0f;
  for (int k = 0; k < N_STATE; ++k) s = fmaf(Wbs[r * 256 + k], Tm[k * 64 + c], s);
  Wmid[idx] = s;
}

// ----------------------------------------------------------------------------
// Prep kernel 4: pack a row-major f32 [K][N] matrix into per-lane-contiguous
// f16 WMMA B-fragments: dst[((kt*NT + nt)*32 + lane)*16 + h]
//   lanes 0-15 : N = lane,     K = kt*32 + (h&7) + 16*(h>>3)      (h = 0..15)
//   lanes 16-31: N = lane-16,  K = kt*32 + 8 + (h&7) + 16*(h>>3)
// (ISA 7.12.2 16-bit B-matrix layout.)  One v16h load per lane per fragment.
// ----------------------------------------------------------------------------
__global__ void k_pack_b(const float* __restrict__ src, _Float16* __restrict__ dst,
                         int K, int N) {
  int p = blockIdx.x * blockDim.x + threadIdx.x;
  if (p >= K * N) return;
  int h    = p & 15;
  int lane = (p >> 4) & 31;
  int rem  = p >> 9;
  int NT   = N >> 4;
  int nt   = rem % NT;
  int kt   = rem / NT;
  int k = kt * 32 + ((lane >> 4) << 3) + (h & 7) + ((h >> 3) << 4);
  int n = nt * 16 + (lane & 15);
  dst[p] = (_Float16)src[k * N + n];
}

// ----------------------------------------------------------------------------
// Main fused kernel: 512 threads = 16 waves.
//   wave w -> row group g = w&3 (16 tokens), slice s = w>>2
//   stage A: 4-way split-K (each wave sums 512 of the 2048 K columns)
//   stage B: waves with s==0 reduce partials + apply W_mid
//   stage C: 4-way split-N (each wave writes 512 of the 2048 output columns)
// ----------------------------------------------------------------------------
__device__ __forceinline__ v8f wmma16(v16h a, v16h b, v8f c) {
  return __builtin_amdgcn_wmma_f32_16x16x32_f16(false, a, false, b, (short)0, c,
                                                false, false);
}
__device__ __forceinline__ v8f v8f_zero() {
  v8f z;
#pragma unroll
  for (int i = 0; i < 8; ++i) z[i] = 0.0f;
  return z;
}

__global__ __launch_bounds__(512) void k_ssm_main(
    const float* __restrict__ u,
    const _Float16* __restrict__ pWb,    // packed W_b_rank  [2048x64]
    const _Float16* __restrict__ pWmid,  // packed W_mid     [64x64]
    const _Float16* __restrict__ pWc,    // packed W_c_model [64x2048]
    const float* __restrict__ Dv,
    float* __restrict__ out) {
  __shared__ float    ldsP[12 * 16 * 64];  // f32 t1 partials from K-slices 1..3
  __shared__ _Float16 ldsT[4 * 16 * 64];   // f16 t1/t2 strip per row group

  const int lane  = threadIdx.x & 31;
  const int wv    = threadIdx.x >> 5;   // 0..15
  const int g     = wv & 3;             // row group (16 tokens)
  const int sl    = wv >> 2;            // K-slice (stage A) / N-slice (stage C)
  const int nLane = lane & 15;          // A: M row / B,C,D: N col
  const int hi    = lane >> 4;          // half-wave select
  const int kadd  = hi * 8;             // A-fragment K offset (upper half-wave)
  const int madd  = hi * 8;             // C/D-fragment M offset (upper half-wave)

  const long tokBase = (long)blockIdx.x * TM;
  const float* up = u + (tokBase + g * 16 + nLane) * (long)D_MODEL;

  const v16h* Wb   = (const v16h*)pWb;
  const v16h* Wmid = (const v16h*)pWmid;
  const v16h* Wc   = (const v16h*)pWc;

  // ---------- stage A (split-K): partial t1[16x64] over kt slice ----------
  v8f acc[4];
#pragma unroll
  for (int nt = 0; nt < 4; ++nt) acc[nt] = v8f_zero();

  const int ktEnd = sl * 16 + 16;
  for (int kt = sl * 16; kt < ktEnd; ++kt) {
    const int k0 = kt * 32 + kadd;
    // A fragment: lanes 0-15 hold K = k0..k0+7 and k0+16..k0+23 of row nLane
    float4 f0 = *(const float4*)(up + k0);
    float4 f1 = *(const float4*)(up + k0 + 4);
    float4 f2 = *(const float4*)(up + k0 + 16);
    float4 f3 = *(const float4*)(up + k0 + 20);
    v16h a;
    a[0]  = (_Float16)f0.x; a[1]  = (_Float16)f0.y; a[2]  = (_Float16)f0.z; a[3]  = (_Float16)f0.w;
    a[4]  = (_Float16)f1.x; a[5]  = (_Float16)f1.y; a[6]  = (_Float16)f1.z; a[7]  = (_Float16)f1.w;
    a[8]  = (_Float16)f2.x; a[9]  = (_Float16)f2.y; a[10] = (_Float16)f2.z; a[11] = (_Float16)f2.w;
    a[12] = (_Float16)f3.x; a[13] = (_Float16)f3.y; a[14] = (_Float16)f3.z; a[15] = (_Float16)f3.w;
#pragma unroll
    for (int nt = 0; nt < 4; ++nt) {
      v16h b = Wb[(kt * 4 + nt) * 32 + lane];
      acc[nt] = wmma16(a, b, acc[nt]);
    }
  }

  // K-slices 1..3 publish f32 partials; slice 0 keeps its partial in registers
  if (sl != 0) {
    float* lp = ldsP + ((sl - 1) * 4 + g) * 1024;
#pragma unroll
    for (int nt = 0; nt < 4; ++nt)
#pragma unroll
      for (int r = 0; r < 8; ++r)
        lp[(r + madd) * 64 + nt * 16 + nLane] = acc[nt][r];
  }
  __syncthreads();

  // ---------- stage B (waves sl==0 only; wave-uniform branch) ----------
  if (sl == 0) {
    _Float16* lt = ldsT + g * 1024;
#pragma unroll
    for (int nt = 0; nt < 4; ++nt)
#pragma unroll
      for (int r = 0; r < 8; ++r) {
        const int off = (r + madd) * 64 + nt * 16 + nLane;
        float s = acc[nt][r];
#pragma unroll
        for (int p = 0; p < 3; ++p) s += ldsP[(p * 4 + g) * 1024 + off];
        lt[off] = (_Float16)s;   // t1 in f16, D-layout positions
      }

    // t1 (LDS) -> A-layout fragments
    v16h a2[2];
#pragma unroll
    for (int kt = 0; kt < 2; ++kt) {
      v8h g0 = *(const v8h*)(lt + nLane * 64 + kt * 32 + kadd);
      v8h g1 = *(const v8h*)(lt + nLane * 64 + kt * 32 + 16 + kadd);
#pragma unroll
      for (int j = 0; j < 8; ++j) { a2[kt][j] = g0[j]; a2[kt][8 + j] = g1[j]; }
    }
    v8f accB[4];
#pragma unroll
    for (int nt = 0; nt < 4; ++nt) accB[nt] = v8f_zero();
#pragma unroll
    for (int kt = 0; kt < 2; ++kt)
#pragma unroll
      for (int nt = 0; nt < 4; ++nt)
        accB[nt] = wmma16(a2[kt], Wmid[(kt * 4 + nt) * 32 + lane], accB[nt]);

    // t2 -> LDS (overwrites t1; t1 reads already landed in registers)
#pragma unroll
    for (int nt = 0; nt < 4; ++nt)
#pragma unroll
      for (int r = 0; r < 8; ++r)
        lt[(r + madd) * 64 + nt * 16 + nLane] = (_Float16)accB[nt][r];
  }
  __syncthreads();

  // ---------- stage C (split-N): out = t2 @ W_c_model + D ⊙ u ----------
  const _Float16* lt = ldsT + g * 1024;
  v16h a3[2];
#pragma unroll
  for (int kt = 0; kt < 2; ++kt) {
    v8h g0 = *(const v8h*)(lt + nLane * 64 + kt * 32 + kadd);
    v8h g1 = *(const v8h*)(lt + nLane * 64 + kt * 32 + 16 + kadd);
#pragma unroll
    for (int j = 0; j < 8; ++j) { a3[kt][j] = g0[j]; a3[kt][8 + j] = g1[j]; }
  }

  float*       outBase = out + (tokBase + g * 16) * (long)D_MODEL;
  const float* uBase   = u   + (tokBase + g * 16) * (long)D_MODEL;
  for (int ntl = 0; ntl < 32; ++ntl) {
    const int nt = sl * 32 + ntl;
    v8f c = v8f_zero();
    c = wmma16(a3[0], Wc[(0 * 128 + nt) * 32 + lane], c);
    c = wmma16(a3[1], Wc[(1 * 128 + nt) * 32 + lane], c);
    const int col = nt * 16 + nLane;
    const float dv = Dv[col];
#pragma unroll
    for (int r = 0; r < 8; ++r) {
      const long off = (long)(r + madd) * D_MODEL + col;
      outBase[off] = c[r] + dv * uBase[off];
    }
  }
}

// ----------------------------------------------------------------------------
extern "C" void kernel_launch(void* const* d_in, const int* in_sizes, int n_in,
                              void* d_out, int out_size, void* d_ws, size_t ws_size,
                              hipStream_t stream) {
  const float* u   = (const float*)d_in[0];
  const float* Af  = (const float*)d_in[1];
  const float* Wbr = (const float*)d_in[2];
  const float* Wbs = (const float*)d_in[3];
  const float* Wcr = (const float*)d_in[4];
  const float* Wcm = (const float*)d_in[5];
  const float* Dv  = (const float*)d_in[6];
  float* out = (float*)d_out;

  // Workspace layout (876544 bytes total, all offsets 32B-aligned)
  char* ws = (char*)d_ws;
  float*    M     = (float*)(ws + 0);        // 256*256*4 = 262144
  float*    Tm    = (float*)(ws + 262144);   // 256*64*4  =  65536
  float*    Wmid  = (float*)(ws + 327680);   // 64*64*4   =  16384
  _Float16* pWb   = (_Float16*)(ws + 344064);// 2048*64*2 = 262144
  _Float16* pWmid = (_Float16*)(ws + 606208);// 64*64*2   =   8192
  _Float16* pWc   = (_Float16*)(ws + 614400);// 64*2048*2 = 262144

  k_butterfly_M<<<dim3(256), dim3(256), 0, stream>>>(Af, M);
  k_m_wcrank   <<<dim3(64),  dim3(256), 0, stream>>>(M, Wcr, Tm);
  k_wbstate_tm <<<dim3(16),  dim3(256), 0, stream>>>(Wbs, Tm, Wmid);
  k_pack_b     <<<dim3(512), dim3(256), 0, stream>>>(Wbr,  pWb,   2048, 64);
  k_pack_b     <<<dim3(16),  dim3(256), 0, stream>>>(Wmid, pWmid, 64,   64);
  k_pack_b     <<<dim3(512), dim3(256), 0, stream>>>(Wcm,  pWc,   64,   2048);

  k_ssm_main<<<dim3(TOKENS / TM), dim3(512), 0, stream>>>(u, pWb, pWmid, pWc, Dv, out);
}
